// MultiLatentAttention_33191507263684
// MI455X (gfx1250) — compile-verified, matrix operand order checked by reading, not run on hardware
//
#include <hip/hip_runtime.h>
#include <cstdint>
#include <cstddef>

// ---------------------------------------------------------------------------
// MLA (Multi-Latent Attention) for MI455X / gfx1250, wave32 + WMMA bf16
// + async global->LDS staging (ASYNCcnt path) with double buffering.
// B=2, S=2048, D=2048, H=16, DK=DV=128, L=64.
// ---------------------------------------------------------------------------

typedef __bf16 bf16_t;
typedef bf16_t v16bf __attribute__((ext_vector_type(16)));
typedef float  v8f   __attribute__((ext_vector_type(8)));

#define MLA_B   2
#define MLA_S   2048
#define MLA_D   2048
#define MLA_H   16
#define MLA_DK  128
#define MLA_DV  128
#define MLA_L   64

// ---------------- async global -> LDS staging -------
#if __has_builtin(__builtin_amdgcn_global_load_async_to_lds_b128)
#define MLA_ASYNC_LDS 1
#else
#define MLA_ASYNC_LDS 0
#endif

// Builtin parameter types: (v4i addrspace(1)*, v4i addrspace(3)*, imm offset, imm cpol)
typedef int v4i_vs __attribute__((vector_size(16)));
typedef __attribute__((address_space(1))) v4i_vs as1_v4i;
typedef __attribute__((address_space(3))) v4i_vs as3_v4i;

// Copy 16 bytes global -> LDS.  Async (one GLOBAL_LOAD_ASYNC_TO_LDS_B128,
// tracked by ASYNCcnt) when available; otherwise VGPR round trip.
__device__ __forceinline__ void stage_b128(const void* g, void* l) {
#if MLA_ASYNC_LDS
  __builtin_amdgcn_global_load_async_to_lds_b128(
      (as1_v4i*)(uintptr_t)g, (as3_v4i*)(uint32_t)(uintptr_t)l, 0, 0);
#else
  *(uint4*)l = *(const uint4*)g;
#endif
}

__device__ __forceinline__ void stage_wait() {
#if MLA_ASYNC_LDS
#if __has_builtin(__builtin_amdgcn_s_wait_asynccnt)
  __builtin_amdgcn_s_wait_asynccnt(0);
#else
  asm volatile("s_wait_asynccnt 0x0" ::: "memory");
#endif
#endif
}

// ---------------- bf16 helpers (RNE) ----------------
__device__ __forceinline__ bf16_t f2bf(float f) {
  union { float f; unsigned u; } v; v.f = f;
  unsigned r = v.u + 0x7FFFu + ((v.u >> 16) & 1u);
  union { unsigned short s; bf16_t b; } o;
  o.s = (unsigned short)(r >> 16);
  return o.b;
}

// ---------------- WMMA fragment loaders -------------
// A-layout (16x32 bf16, M x K), row-major source, leading dim `ld` (elements).
// Lane 0-15: row=lane, K = {0..7, 16..23}; lane 16-31: row=lane-16, K = {8..15, 24..31}.
__device__ __forceinline__ v16bf load_frag_row(const bf16_t* __restrict__ src, int ld) {
  const int lane = threadIdx.x & 31;
  const int row  = lane & 15;
  const int kb   = (lane >> 4) << 3;
  const bf16_t* p = src + (size_t)row * ld + kb;
  v16bf f;
#pragma unroll
  for (int i = 0; i < 8; ++i) f[i] = p[i];
#pragma unroll
  for (int i = 0; i < 8; ++i) f[8 + i] = p[16 + i];
  return f;
}

// B-layout fragment from a K-major source: element (k,n) at src[k*ld + n].
// Used for V in attention (O += P @ V).
__device__ __forceinline__ v16bf load_frag_col(const bf16_t* __restrict__ src, int ld) {
  const int lane = threadIdx.x & 31;
  const int n    = lane & 15;
  const int kb   = (lane >> 4) << 3;
  v16bf f;
#pragma unroll
  for (int i = 0; i < 8; ++i) f[i]     = src[(size_t)(kb + i) * ld + n];
#pragma unroll
  for (int i = 0; i < 8; ++i) f[8 + i] = src[(size_t)(kb + 16 + i) * ld + n];
  return f;
}

__device__ __forceinline__ v8f wmma_bf16(v16bf a, v16bf b, v8f c) {
  return __builtin_amdgcn_wmma_f32_16x16x32_bf16(false, a, false, b, (short)0, c,
                                                 false, false);
}

// ---------------- f32 -> bf16 conversion ------------
__global__ __launch_bounds__(256)
void cvt_f32_bf16(const float* __restrict__ x, bf16_t* __restrict__ y, size_t n) {
  size_t i = (size_t)blockIdx.x * blockDim.x + threadIdx.x;
  size_t stride = (size_t)gridDim.x * blockDim.x;
  for (; i < n; i += stride) y[i] = f2bf(x[i]);
}

// ---------------- generic WMMA GEMM -----------------
// C[M,N] = A[M,K] * B[N,K]^T + bias[N];  A,B bf16 row-major; C f32 or bf16.
// Block: 256 threads = 8 waves; tile 256(M) x 128(N); K stepped by 32 through
// double-buffered LDS staged with async global->LDS copies.
// Wave (wm 0..3, wn 0..1) owns a 64x64 region = 4x4 WMMA tiles
// (16 WMMAs per stage from 8 LDS fragment loads).
__global__ __launch_bounds__(256)
void gemm_wmma_bf16(const bf16_t* __restrict__ A, int lda,
                    const bf16_t* __restrict__ B, int ldb,
                    const float* __restrict__ bias,
                    void* __restrict__ Cout, int ldc,
                    int K, int c_is_f32) {
  __shared__ __align__(16) bf16_t sA[2][256][32];
  __shared__ __align__(16) bf16_t sB[2][128][32];

  const int tid  = threadIdx.x;
  const int wave = tid >> 5;
  const int wm   = wave >> 1;  // 0..3 -> M offset wm*64
  const int wn   = wave & 1;   // 0..1 -> N offset wn*64
  const int bm   = blockIdx.y * 256;
  const int bn   = blockIdx.x * 128;

  auto stage = [&](int p, int k0) {
#pragma unroll
    for (int c = 0; c < 4; ++c) {            // A: 256x32 = 1024 chunks of 8
      const int chunk = tid + c * 256;
      const int row   = chunk >> 2;
      const int col   = (chunk & 3) << 3;
      stage_b128(&A[(size_t)(bm + row) * lda + k0 + col], &sA[p][row][col]);
    }
#pragma unroll
    for (int c = 0; c < 2; ++c) {            // B: 128x32 = 512 chunks of 8
      const int chunk = tid + c * 256;
      const int row   = chunk >> 2;
      const int col   = (chunk & 3) << 3;
      stage_b128(&B[(size_t)(bn + row) * ldb + k0 + col], &sB[p][row][col]);
    }
  };

  v8f acc[4][4];
#pragma unroll
  for (int i = 0; i < 4; ++i)
#pragma unroll
    for (int j = 0; j < 4; ++j)
#pragma unroll
      for (int r = 0; r < 8; ++r) acc[i][j][r] = 0.0f;

  const int nk = K >> 5;
  stage(0, 0);
  stage_wait();
  __syncthreads();

  for (int i = 0; i < nk; ++i) {
    const int p = i & 1;
    if (i + 1 < nk) stage(p ^ 1, (i + 1) * 32);   // overlap with WMMAs below
    if (i + 2 < nk) {                             // L2 hint two stages ahead
      __builtin_prefetch(&A[(size_t)(bm + (tid >> 1)) * lda + (i + 2) * 32], 0, 1);
      __builtin_prefetch(&B[(size_t)(bn + (tid >> 2)) * ldb + (i + 2) * 32], 0, 1);
    }

    v16bf a[4];
#pragma unroll
    for (int mi = 0; mi < 4; ++mi)
      a[mi] = load_frag_row(&sA[p][wm * 64 + mi * 16][0], 32);
#pragma unroll
    for (int nj = 0; nj < 4; ++nj) {
      v16bf b = load_frag_row(&sB[p][wn * 64 + nj * 16][0], 32);
#pragma unroll
      for (int mi = 0; mi < 4; ++mi)
        acc[mi][nj] = wmma_bf16(a[mi], b, acc[mi][nj]);
    }

    stage_wait();     // next buffer landed (and our issues retired)
    __syncthreads();  // everyone done with buffer p; data visible block-wide
  }

  // Epilogue: C layout = VGPR r -> row (r + 8*(lane>=16)), lanes 0..15 -> col.
  const int lane = tid & 31;
  const int cn   = lane & 15;
  const int rb   = (lane >> 4) << 3;
#pragma unroll
  for (int mi = 0; mi < 4; ++mi)
#pragma unroll
    for (int nj = 0; nj < 4; ++nj) {
      const int col = bn + wn * 64 + nj * 16 + cn;
      const float bv = bias ? bias[col] : 0.0f;
#pragma unroll
      for (int r = 0; r < 8; ++r) {
        const int row = bm + wm * 64 + mi * 16 + rb + r;
        const float v = acc[mi][nj][r] + bv;
        if (c_is_f32)
          ((float*)Cout)[(size_t)row * ldc + col] = v;
        else
          ((bf16_t*)Cout)[(size_t)row * ldc + col] = f2bf(v);
      }
    }
}

// ---------------- flash attention (per b,h) ---------
// Q,K,V stored [B*S, H*128] bf16; O written [B*S, H*128] bf16.
// Block = 128 threads (4 waves); 64 query rows per block (wave owns 16),
// streaming 64-key tiles with online softmax; causal mask.
__global__ __launch_bounds__(128)
void mla_flash_attn(const bf16_t* __restrict__ Q, const bf16_t* __restrict__ Kr,
                    const bf16_t* __restrict__ Vr, bf16_t* __restrict__ O) {
  const int h    = blockIdx.y & (MLA_H - 1);
  const int b    = blockIdx.y >> 4;
  const int qt   = blockIdx.x;         // 64-row query tile index
  const int tid  = threadIdx.x;
  const int wave = tid >> 5;
  const int lane = tid & 31;
  const int ld   = MLA_H * MLA_DK;     // 2048

  __shared__ __align__(16) bf16_t sQ[64][128];
  __shared__ __align__(16) bf16_t sK[64][128];
  __shared__ __align__(16) bf16_t sV[64][128];
  __shared__ __align__(16) bf16_t sP[4][16][64];

  const bf16_t* qbase = Q + ((size_t)b * MLA_S + (size_t)qt * 64) * ld + h * MLA_DK;
#pragma unroll
  for (int c = 0; c < 8; ++c) {        // 64x128 bf16 in 8 rounds of b128
    const int chunk = tid + c * 128;
    const int row   = chunk >> 4;
    const int col   = (chunk & 15) << 3;
    stage_b128(&qbase[(size_t)row * ld + col], &sQ[row][col]);
  }

  v8f acco[8];
  float mrow[8], lrow[8];
#pragma unroll
  for (int n = 0; n < 8; ++n)
#pragma unroll
    for (int r = 0; r < 8; ++r) acco[n][r] = 0.0f;
#pragma unroll
  for (int r = 0; r < 8; ++r) { mrow[r] = -3.0e38f; lrow[r] = 0.0f; }

  const float scale = 0.08838834764831845f;  // 1/sqrt(128)
  const int cn = lane & 15;
  const int rb = (lane >> 4) << 3;

  for (int j = 0; j <= qt; ++j) {
    __syncthreads();  // retire prior reads of sK/sV
    const bf16_t* kbase = Kr + ((size_t)b * MLA_S + (size_t)j * 64) * ld + h * MLA_DK;
    const bf16_t* vbase = Vr + ((size_t)b * MLA_S + (size_t)j * 64) * ld + h * MLA_DV;
#pragma unroll
    for (int c = 0; c < 8; ++c) {
      const int chunk = tid + c * 128;
      const int row   = chunk >> 4;
      const int col   = (chunk & 15) << 3;
      stage_b128(&kbase[(size_t)row * ld + col], &sK[row][col]);
      stage_b128(&vbase[(size_t)row * ld + col], &sV[row][col]);
    }
    stage_wait();     // covers Q staging on the first iteration too
    __syncthreads();

    // S = Q * K^T for this wave's 16 query rows (16x64 in 4 C tiles).
    v8f accs[4];
#pragma unroll
    for (int n = 0; n < 4; ++n)
#pragma unroll
      for (int r = 0; r < 8; ++r) accs[n][r] = 0.0f;
#pragma unroll
    for (int kt = 0; kt < 4; ++kt) {
      v16bf a = load_frag_row(&sQ[wave * 16][kt * 32], 128);
#pragma unroll
      for (int n = 0; n < 4; ++n) {
        v16bf bf = load_frag_row(&sK[n * 16][kt * 32], 128);
        accs[n] = wmma_bf16(a, bf, accs[n]);
      }
    }

    // Scale + causal mask (only the diagonal tile needs masking).
    const int q0 = qt * 64 + wave * 16;
    if (j == qt) {
#pragma unroll
      for (int n = 0; n < 4; ++n) {
        const int key = j * 64 + n * 16 + cn;
#pragma unroll
        for (int r = 0; r < 8; ++r) {
          const int qi = q0 + rb + r;
          accs[n][r] = (key > qi) ? -3.0e38f : accs[n][r] * scale;
        }
      }
    } else {
#pragma unroll
      for (int n = 0; n < 4; ++n)
#pragma unroll
        for (int r = 0; r < 8; ++r) accs[n][r] *= scale;
    }

    // Online softmax: per-row max / exp / sum (16-lane row groups).
#pragma unroll
    for (int r = 0; r < 8; ++r) {
      float m = fmaxf(fmaxf(accs[0][r], accs[1][r]),
                      fmaxf(accs[2][r], accs[3][r]));
      m = fmaxf(m, __shfl_xor(m, 1, 32));
      m = fmaxf(m, __shfl_xor(m, 2, 32));
      m = fmaxf(m, __shfl_xor(m, 4, 32));
      m = fmaxf(m, __shfl_xor(m, 8, 32));
      const float mnew = fmaxf(mrow[r], m);
      float s = 0.0f;
#pragma unroll
      for (int n = 0; n < 4; ++n) {
        const float p = __expf(accs[n][r] - mnew);
        accs[n][r] = p;
        s += p;
      }
      s += __shfl_xor(s, 1, 32);
      s += __shfl_xor(s, 2, 32);
      s += __shfl_xor(s, 4, 32);
      s += __shfl_xor(s, 8, 32);
      const float corr = __expf(mrow[r] - mnew);
      lrow[r] = lrow[r] * corr + s;
      mrow[r] = mnew;
#pragma unroll
      for (int n = 0; n < 8; ++n) acco[n][r] *= corr;
    }

    // Re-layout P (C layout -> A layout) through LDS, as bf16.
#pragma unroll
    for (int n = 0; n < 4; ++n)
#pragma unroll
      for (int r = 0; r < 8; ++r)
        sP[wave][rb + r][n * 16 + cn] = f2bf(accs[n][r]);

    // O += P(16x64) @ V(64x128).
#pragma unroll
    for (int kt = 0; kt < 2; ++kt) {
      v16bf a = load_frag_row(&sP[wave][0][kt * 32], 64);
#pragma unroll
      for (int n = 0; n < 8; ++n) {
        v16bf bf = load_frag_col(&sV[kt * 32][n * 16], 128);
        acco[n] = wmma_bf16(a, bf, acco[n]);
      }
    }
  }

  // Finalize: divide by row sums, store bf16 into [B*S, H*DV].
  bf16_t* obase =
      O + ((size_t)b * MLA_S + (size_t)qt * 64 + wave * 16) * (MLA_H * MLA_DV) +
      h * MLA_DV;
#pragma unroll
  for (int n = 0; n < 8; ++n)
#pragma unroll
    for (int r = 0; r < 8; ++r) {
      const float v = acco[n][r] / lrow[r];
      obase[(size_t)(rb + r) * (MLA_H * MLA_DV) + n * 16 + cn] = f2bf(v);
    }
}

// ---------------------------------------------------------------------------
// Host launcher
// ---------------------------------------------------------------------------
static inline size_t align256(size_t x) { return (x + 255) & ~(size_t)255; }

extern "C" void kernel_launch(void* const* d_in, const int* in_sizes, int n_in,
                              void* d_out, int out_size, void* d_ws, size_t ws_size,
                              hipStream_t stream) {
  (void)in_sizes; (void)n_in; (void)out_size; (void)ws_size;
  const float* queries = (const float*)d_in[0];
  const float* keys    = (const float*)d_in[1];
  const float* values  = (const float*)d_in[2];
  const float* Wq      = (const float*)d_in[3];
  const float* bq      = (const float*)d_in[4];
  const float* Wlk     = (const float*)d_in[5];
  const float* blk_b   = (const float*)d_in[6];
  const float* Wlv     = (const float*)d_in[7];
  const float* blv     = (const float*)d_in[8];
  const float* Wkr     = (const float*)d_in[9];
  const float* Wvr     = (const float*)d_in[10];
  const float* Wo      = (const float*)d_in[11];
  const float* bo      = (const float*)d_in[12];

  const size_t BS   = (size_t)MLA_B * MLA_S;          // 4096
  const size_t nAct = BS * MLA_D;                     // 8.39M
  const size_t nQ   = BS * MLA_H * MLA_DK;            // 8.39M
  const size_t nLK  = BS * MLA_H * MLA_L;             // 4.19M

  char* w = (char*)d_ws;
  auto carve = [&](size_t elems) {
    bf16_t* p = (bf16_t*)w;
    w += align256(elems * sizeof(bf16_t));
    return p;
  };
  bf16_t* xq   = carve(nAct);
  bf16_t* xk   = carve(nAct);
  bf16_t* xv   = carve(nAct);
  bf16_t* wqb  = carve((size_t)MLA_H * MLA_DK * MLA_D);
  bf16_t* wlkb = carve((size_t)MLA_H * MLA_L * MLA_D);
  bf16_t* wlvb = carve((size_t)MLA_H * MLA_L * MLA_D);
  bf16_t* wkrb = carve((size_t)MLA_DK * MLA_L);
  bf16_t* wvrb = carve((size_t)MLA_DV * MLA_L);
  bf16_t* wob  = carve((size_t)MLA_D * MLA_H * MLA_DV);
  bf16_t* qb   = carve(nQ);
  bf16_t* lkb  = carve(nLK);
  bf16_t* lvb  = carve(nLK);
  bf16_t* kb   = carve(nQ);
  bf16_t* vb   = carve(nQ);
  bf16_t* aob  = carve(nQ);

  const dim3 cb(256);
  const int  cgrid = 2048;
  hipLaunchKernelGGL(cvt_f32_bf16, dim3(cgrid), cb, 0, stream, queries, xq, nAct);
  hipLaunchKernelGGL(cvt_f32_bf16, dim3(cgrid), cb, 0, stream, keys,    xk, nAct);
  hipLaunchKernelGGL(cvt_f32_bf16, dim3(cgrid), cb, 0, stream, values,  xv, nAct);
  hipLaunchKernelGGL(cvt_f32_bf16, dim3(cgrid), cb, 0, stream, Wq,  wqb,  (size_t)MLA_H * MLA_DK * MLA_D);
  hipLaunchKernelGGL(cvt_f32_bf16, dim3(cgrid), cb, 0, stream, Wlk, wlkb, (size_t)MLA_H * MLA_L * MLA_D);
  hipLaunchKernelGGL(cvt_f32_bf16, dim3(cgrid), cb, 0, stream, Wlv, wlvb, (size_t)MLA_H * MLA_L * MLA_D);
  hipLaunchKernelGGL(cvt_f32_bf16, dim3(32),    cb, 0, stream, Wkr, wkrb, (size_t)MLA_DK * MLA_L);
  hipLaunchKernelGGL(cvt_f32_bf16, dim3(32),    cb, 0, stream, Wvr, wvrb, (size_t)MLA_DV * MLA_L);
  hipLaunchKernelGGL(cvt_f32_bf16, dim3(cgrid), cb, 0, stream, Wo,  wob,  (size_t)MLA_D * MLA_H * MLA_DV);

  const dim3 gb(256);
  // q = queries @ Wq.T + bq            -> [4096, 2048] bf16
  hipLaunchKernelGGL(gemm_wmma_bf16, dim3(2048 / 128, 4096 / 256), gb, 0, stream,
                     xq, MLA_D, wqb, MLA_D, bq, (void*)qb, MLA_H * MLA_DK, MLA_D, 0);
  // lk = keys @ Wlk.T + blk            -> [4096, 1024] bf16
  hipLaunchKernelGGL(gemm_wmma_bf16, dim3(1024 / 128, 4096 / 256), gb, 0, stream,
                     xk, MLA_D, wlkb, MLA_D, blk_b, (void*)lkb, MLA_H * MLA_L, MLA_D, 0);
  // lv = values @ Wlv.T + blv          -> [4096, 1024] bf16
  hipLaunchKernelGGL(gemm_wmma_bf16, dim3(1024 / 128, 4096 / 256), gb, 0, stream,
                     xv, MLA_D, wlvb, MLA_D, blv, (void*)lvb, MLA_H * MLA_L, MLA_D, 0);
  // k = lk @ Wkr.T : view lk as [B*S*H, L] -> [65536, 128] bf16
  hipLaunchKernelGGL(gemm_wmma_bf16, dim3(1, 65536 / 256), gb, 0, stream,
                     lkb, MLA_L, wkrb, MLA_L, (const float*)nullptr, (void*)kb,
                     MLA_DK, MLA_L, 0);
  // v = lv @ Wvr.T                      -> [65536, 128] bf16
  hipLaunchKernelGGL(gemm_wmma_bf16, dim3(1, 65536 / 256), gb, 0, stream,
                     lvb, MLA_L, wvrb, MLA_L, (const float*)nullptr, (void*)vb,
                     MLA_DV, MLA_L, 0);
  // causal flash attention per (b,h)   -> attn_out [4096, 2048] bf16
  hipLaunchKernelGGL(mla_flash_attn, dim3(MLA_S / 64, MLA_B * MLA_H), dim3(128), 0,
                     stream, qb, kb, vb, aob);
  // out = attn_out @ Wo.T + bo         -> d_out f32 [4096, 2048]
  hipLaunchKernelGGL(gemm_wmma_bf16, dim3(2048 / 128, 4096 / 256), gb, 0, stream,
                     aob, MLA_H * MLA_DV, wob, MLA_H * MLA_DV, bo, d_out, MLA_D,
                     MLA_H * MLA_DV, 1);
}
